// SparseMoe_74569222193397
// MI455X (gfx1250) — matrix operand order repaired and checked by reading, not compiled, not run on hardware
//
#include <hip/hip_runtime.h>

// ---------------------------------------------------------------------------
// Sparse MoE (top-2 of 8 experts), B=4 S=2048 D=512 H=2048, T=8192 tokens.
// bf16 WMMA (v_wmma_f32_16x16x32_bf16) with fp32 accumulation.
//   * M=32 token tile per WG (2 WMMA M-tiles/wave) -> halves L2 weight traffic
//   * gate score folded into h rows before GEMM2 -> single final accumulator
//   * x tile staged LDS via global_load_async_to_lds_b128 (ASYNCcnt path)
// Workspace: w1t 16.78MB + w2t 16.78MB + xbf 8.39MB + scores 0.26MB ~= 42.3MB
// ---------------------------------------------------------------------------

#define USE_ASYNC_X 1

typedef __attribute__((ext_vector_type(16))) __bf16        v16bf;
typedef __attribute__((ext_vector_type(8)))  float         v8f;
typedef __attribute__((ext_vector_type(4)))  unsigned int  v4u;

union FragB16 { v16bf v; v4u q[2]; };     // 32B: 16 bf16 halves
union Acc8    { v8f   v; float f[8]; };   // 32B: 8 f32 accumulators

static constexpr int kB = 4, kS = 2048, kD = 512, kE = 8, kH = 2048;
static constexpr int kT = kB * kS;        // 8192 tokens
static constexpr int kHC = 128;           // H-chunk per GEMM1/GEMM2 round
static constexpr int kM  = 32;            // tokens per workgroup (2 M-tiles)

__device__ __forceinline__ unsigned short f2bf(float f) {
  unsigned int u = __builtin_bit_cast(unsigned int, f);
  unsigned int r = u + 0x7FFFu + ((u >> 16) & 1u);
  return (unsigned short)(r >> 16);
}

// ---------------------------------------------------------------------------
// Kernel 0a: src [E][P][Q] fp32 -> dst [E][Q][P] bf16 (transpose + convert).
// ---------------------------------------------------------------------------
__global__ void cvt_transpose_bf16(const float* __restrict__ src,
                                   unsigned short* __restrict__ dst,
                                   int P, int Q) {
  const int  per   = P >> 3;
  const long total = (long)kE * Q * per;
  for (long j = (long)blockIdx.x * blockDim.x + threadIdx.x; j < total;
       j += (long)gridDim.x * blockDim.x) {
    const int  e   = (int)(j / ((long)Q * per));
    const long rem = j % ((long)Q * per);
    const int  q   = (int)(rem / per);
    const int  p0  = (int)(rem % per) * 8;
    const float* s = src + ((size_t)e * P + p0) * Q + q;
    unsigned short* d = dst + ((size_t)e * Q + q) * P + p0;
    union { unsigned short h[8]; v4u q; } tw;
#pragma unroll
    for (int i = 0; i < 8; ++i) tw.h[i] = f2bf(s[(size_t)i * Q]);
    *(v4u*)d = tw.q;
  }
}

// ---------------------------------------------------------------------------
// Kernel 0b: contiguous fp32 -> bf16 (for x), 8 elements / thread.
// ---------------------------------------------------------------------------
__global__ void cvt_bf16(const float* __restrict__ src,
                         unsigned short* __restrict__ dst, long n8) {
  for (long j = (long)blockIdx.x * blockDim.x + threadIdx.x; j < n8;
       j += (long)gridDim.x * blockDim.x) {
    const float* s = src + j * 8;
    union { unsigned short h[8]; v4u q; } tw;
#pragma unroll
    for (int i = 0; i < 8; ++i) tw.h[i] = f2bf(s[i]);
    *(v4u*)(dst + j * 8) = tw.q;
  }
}

// ---------------------------------------------------------------------------
// Kernel 1: router. One wave per token; top-k (first-index ties) + sparse
// softmax -> dense scores [T][E].
// ---------------------------------------------------------------------------
__global__ __launch_bounds__(256)
void moe_router(const float* __restrict__ x,
                const float* __restrict__ rw,   // [D][E]
                const float* __restrict__ rb,   // [E]
                const int*   __restrict__ topk_p,
                float* __restrict__ scores) {   // [T][E]
  const int tok  = (blockIdx.x * blockDim.x + threadIdx.x) >> 5;
  const int lane = threadIdx.x & 31;
  if (tok >= kT) return;
  const float* xr = x + (size_t)tok * kD;

  float acc[kE];
#pragma unroll
  for (int e = 0; e < kE; ++e) acc[e] = 0.f;
  for (int d = lane; d < kD; d += 32) {
    const float xv = xr[d];
    const float* w = rw + d * kE;
#pragma unroll
    for (int e = 0; e < kE; ++e) acc[e] = fmaf(xv, w[e], acc[e]);
  }
#pragma unroll
  for (int e = 0; e < kE; ++e) {
    float v = acc[e];
#pragma unroll
    for (int off = 16; off >= 1; off >>= 1) v += __shfl_xor(v, off, 32);
    acc[e] = v + rb[e];
  }

  if (lane == 0) {
    int k = *topk_p;
    if (k < 1) k = 1;
    if (k > kE) k = kE;
    float sv[kE]; bool used[kE];
    for (int e = 0; e < kE; ++e) { sv[e] = 0.f; used[e] = false; }
    float tv[kE]; int ti[kE];
    for (int j = 0; j < k; ++j) {
      float best = -3.402823e38f; int bi = 0;
      for (int e = 0; e < kE; ++e)
        if (!used[e] && acc[e] > best) { best = acc[e]; bi = e; }
      used[bi] = true; tv[j] = best; ti[j] = bi;
    }
    const float m0 = tv[0];
    float sum = 0.f;
    for (int j = 0; j < k; ++j) { tv[j] = __expf(tv[j] - m0); sum += tv[j]; }
    const float inv = 1.f / sum;
    for (int j = 0; j < k; ++j) sv[ti[j]] = tv[j] * inv;
    float* so = scores + (size_t)tok * kE;
    for (int e = 0; e < kE; ++e) so[e] = sv[e];
  }
}

// ---------------------------------------------------------------------------
// Kernel 2: fused expert FFN. WG = 32-token tile (2 M-tiles), 8 waves.
//   GEMM1: h[32,128] = relu(x[32,512] @ w1[:,chunk] + b1); rows scaled by
//          this expert's gate score BEFORE the bf16 store to LDS, so
//   GEMM2 accumulates sum_e (diag(s_e) h_e) @ w2 directly into fin[].
//   Per-expert b2 handled as fin += s_e * b2[e].
// Fragment layouts used here (wave32; A and B both contiguous K, not the ISA table order):
//   A 16x32 bf16 : lane<16 -> M=lane, K k0..k0+15; lane>=16 -> M=lane-16, K k0+16..k0+31
//   B 32x16 bf16 : lane<16 -> N=lane, K k0..k0+15; lane>=16 -> K k0+16..k0+31
//   C/D 16x16 f32: lane<16 -> N=lane, M=v; lane>=16 -> N=lane-16, M=v+8
// ---------------------------------------------------------------------------
__global__ __launch_bounds__(256)
void moe_ffn(const unsigned short* __restrict__ xbf,  // [T][D] bf16
             const unsigned short* __restrict__ w1t,  // [E][H][D] bf16
             const float* __restrict__ b1,            // [E][H]
             const unsigned short* __restrict__ w2t,  // [E][D][H] bf16
             const float* __restrict__ b2,            // [E][D]
             const float* __restrict__ scores,        // [T][E]
             float* __restrict__ out) {               // [T][D]
  __shared__ unsigned short lds_x[kM * kD];    // 32 KB, bf16 x tile
  __shared__ unsigned short lds_h[kM * kHC];   //  8 KB, bf16 gated-h chunk
  __shared__ float          lds_s[kM * kE];    //  1 KB, scores tile

  const int t0  = blockIdx.x * kM;
  const int tid = threadIdx.x;

  // scores tile: exactly 256 values
  lds_s[tid] = scores[(size_t)(t0 + (tid >> 3)) * kE + (tid & 7)];

  // x tile: 32x512 bf16 = 2048 x 16B chunks, 8 per thread
#if USE_ASYNC_X
  {
    const unsigned short* gx = xbf + (size_t)t0 * kD;
#pragma unroll
    for (int it = 0; it < 8; ++it) {
      const int cidx = tid + it * 256;                       // 16B chunk index
      unsigned lds_addr = (unsigned)(size_t)(&lds_x[cidx * 8]);
      const unsigned short* ga = gx + (size_t)cidx * 8;
      asm volatile("global_load_async_to_lds_b128 %0, %1, off"
                   :: "v"(lds_addr), "v"(ga) : "memory");
    }
    asm volatile("s_wait_asynccnt 0x0" ::: "memory");
  }
#else
  for (int i = tid; i < kM * kD / 8; i += 256)
    *(v4u*)&lds_x[i * 8] = *(const v4u*)&xbf[(size_t)t0 * kD + i * 8];
#endif
  __syncthreads();

  const int wave = tid >> 5;
  const int lane = tid & 31;
  const int c    = lane & 15;   // A row / B-C column within 16x16 tile
  const int hi   = lane >> 4;   // K-half selector (and M+8 for C/D)

  Acc8 fin[2][4];
#pragma unroll
  for (int mt = 0; mt < 2; ++mt)
#pragma unroll
    for (int nt = 0; nt < 4; ++nt)
#pragma unroll
      for (int v = 0; v < 8; ++v) fin[mt][nt].f[v] = 0.f;

  for (int e = 0; e < kE; ++e) {
    float sv[2][8];
#pragma unroll
    for (int mt = 0; mt < 2; ++mt)
#pragma unroll
      for (int v = 0; v < 8; ++v)
        sv[mt][v] = lds_s[(mt * 16 + v + hi * 8) * kE + e];

    const unsigned short* w1e = w1t + (size_t)e * kH * kD;
    const unsigned short* w2e = w2t + (size_t)e * kD * kH;

    for (int hc = 0; hc < kH / kHC; ++hc) {
      const int ncol = hc * kHC + wave * 16 + c;   // h column this lane owns
      const unsigned short* w1row = w1e + (size_t)ncol * kD;
      if (hc + 1 < kH / kHC)
        __builtin_prefetch(w1row + (size_t)kHC * kD, 0, 0);

      // ---- GEMM1: two 16x16 h-tiles over K = D = 512, B shared ----
      Acc8 hacc[2];
#pragma unroll
      for (int mt = 0; mt < 2; ++mt)
#pragma unroll
        for (int v = 0; v < 8; ++v) hacc[mt].f[v] = 0.f;
#pragma unroll
      for (int k0 = 0; k0 < kD; k0 += 32) {
        FragB16 b;
        const unsigned short* bp = w1row + k0 + hi * 16;
        b.q[0] = *(const v4u*)bp;
        b.q[1] = *(const v4u*)(bp + 8);
#pragma unroll
        for (int mt = 0; mt < 2; ++mt) {
          FragB16 a;
          const unsigned short* ap = &lds_x[(mt * 16 + c) * kD + k0 + hi * 16];
          a.q[0] = *(const v4u*)ap;
          a.q[1] = *(const v4u*)(ap + 8);
          hacc[mt].v = __builtin_amdgcn_wmma_f32_16x16x32_bf16(
              false, a.v, false, b.v, (short)0, hacc[mt].v, false, false);
        }
      }

      const float b1v = b1[(size_t)e * kH + ncol];
      __syncthreads();                        // prior chunk's GEMM2 reads done
#pragma unroll
      for (int mt = 0; mt < 2; ++mt)
#pragma unroll
        for (int v = 0; v < 8; ++v) {
          float hv = hacc[mt].f[v] + b1v;
          hv = hv > 0.f ? hv : 0.f;           // ReLU
          hv *= sv[mt][v];                    // fold gate into h rows
          lds_h[(mt * 16 + v + hi * 8) * kHC + wave * 16 + c] = f2bf(hv);
        }
      __syncthreads();                        // gated h visible to all waves

      // ---- GEMM2: fin[mt][nt] += h_chunk @ w2[chunk,:], B shared over mt ----
#pragma unroll
      for (int nt = 0; nt < 4; ++nt) {
        const unsigned short* w2row =
            w2e + (size_t)(wave * 64 + nt * 16 + c) * kH + hc * kHC;
#pragma unroll
        for (int kk = 0; kk < kHC; kk += 32) {
          FragB16 b;
          const unsigned short* bp = w2row + kk + hi * 16;
          b.q[0] = *(const v4u*)bp;
          b.q[1] = *(const v4u*)(bp + 8);
#pragma unroll
          for (int mt = 0; mt < 2; ++mt) {
            FragB16 a;
            const unsigned short* ap = &lds_h[(mt * 16 + c) * kHC + kk + hi * 16];
            a.q[0] = *(const v4u*)ap;
            a.q[1] = *(const v4u*)(ap + 8);
            fin[mt][nt].v = __builtin_amdgcn_wmma_f32_16x16x32_bf16(
                false, a.v, false, b.v, (short)0, fin[mt][nt].v, false, false);
          }
        }
      }
    }

    // fin += s_e * b2[e]  (the gated bias term of this expert)
#pragma unroll
    for (int nt = 0; nt < 4; ++nt) {
      const float b2v = b2[(size_t)e * kD + wave * 64 + nt * 16 + c];
#pragma unroll
      for (int mt = 0; mt < 2; ++mt)
#pragma unroll
        for (int v = 0; v < 8; ++v)
          fin[mt][nt].f[v] = fmaf(sv[mt][v], b2v, fin[mt][nt].f[v]);
    }
  }

#pragma unroll
  for (int mt = 0; mt < 2; ++mt)
#pragma unroll
    for (int nt = 0; nt < 4; ++nt) {
      const int col = wave * 64 + nt * 16 + c;
#pragma unroll
      for (int v = 0; v < 8; ++v)
        out[(size_t)(t0 + mt * 16 + v + hi * 8) * kD + col] = fin[mt][nt].f[v];
    }
}

// ---------------------------------------------------------------------------
extern "C" void kernel_launch(void* const* d_in, const int* in_sizes, int n_in,
                              void* d_out, int out_size, void* d_ws, size_t ws_size,
                              hipStream_t stream) {
  const float* x   = (const float*)d_in[0];
  const float* rw  = (const float*)d_in[1];
  const float* rb  = (const float*)d_in[2];
  const float* w1  = (const float*)d_in[3];
  const float* b1  = (const float*)d_in[4];
  const float* w2  = (const float*)d_in[5];
  const float* b2  = (const float*)d_in[6];
  const int* topk  = (const int*)d_in[7];
  float* out       = (float*)d_out;

  // workspace layout (all 16B aligned)
  char* ws = (char*)d_ws;
  size_t off = 0;
  unsigned short* w1t = (unsigned short*)(ws + off); off += (size_t)kE * kH * kD * 2;
  unsigned short* w2t = (unsigned short*)(ws + off); off += (size_t)kE * kD * kH * 2;
  unsigned short* xbf = (unsigned short*)(ws + off); off += (size_t)kT * kD * 2;
  float* scores       = (float*)(ws + off);
  (void)in_sizes; (void)n_in; (void)out_size; (void)ws_size;

  // 0) convert: w1[E][D][H]->w1t[E][H][D], w2[E][H][D]->w2t[E][D][H], x->xbf
  cvt_transpose_bf16<<<4096, 256, 0, stream>>>(w1, w1t, kD, kH);
  cvt_transpose_bf16<<<4096, 256, 0, stream>>>(w2, w2t, kH, kD);
  cvt_bf16<<<2048, 256, 0, stream>>>(x, xbf, (long)kT * kD / 8);
  // 1) router -> dense scores [T][E]
  moe_router<<<kT / 8, 256, 0, stream>>>(x, rw, rb, topk, scores);
  // 2) fused expert FFN + gated combine (256 WGs x 8 waves)
  moe_ffn<<<kT / kM, 256, 0, stream>>>(xbf, w1t, b1, w2t, b2, scores, out);
}
